// DuelingQNetwork_36120674959730
// MI455X (gfx1250) — matrix operand order repaired and checked by reading, not compiled, not run on hardware
//
#include <hip/hip_runtime.h>
#include <hip/hip_bf16.h>

typedef _Float16 v16h __attribute__((ext_vector_type(16)));
typedef _Float16 v8h  __attribute__((ext_vector_type(8)));
typedef float    v8f  __attribute__((ext_vector_type(8)));

#define WMMA(a, b, c) __builtin_amdgcn_wmma_f32_16x16x32_f16( \
    false, (a), false, (b), (short)0, (c), false, false)

// ---- weight LDS offsets (halves). Column-major, K padded to mult of 32,
// N padded to 16 for the N=1 heads. All offsets 32B aligned (mult of 16).
#define OWD1  0u       // (K4 ->32) x N32    = 1024
#define OWD2  1024u    // (K32)     x N32    = 1024
#define OWT1  2048u    // (K18->32) x N32    = 1024
#define OWT2  3072u    // (K32)     x N32    = 1024
#define OWV1  4096u    // (K53->64) x N64    = 4096
#define OWV2  8192u    // (K64)     x N32    = 2048
#define OWV3  10240u   // (K32)     x N1->16 =  512
#define OWS1  10752u   // (K53->64) x N32    = 2048
#define OWS2  12800u   // (K32)     x N1->16 =  512
#define OWP1A 13312u   // wp1 rows  0..31 (K32)     x N64 = 2048
#define OWP1B 15360u   // wp1 rows 32..84 (K53->64) x N64 = 4096
#define OWP2  19456u   // (K64)     x N16    = 1024
#define OWP3  20480u   // (K16->32) x N1->16 =  512
#define TOTW  20992u

#define TILES 8   // 16-sample tiles per wave; amortizes weight staging

struct Params {
  const float* state; const float* emb;
  const float* wd1; const float* bd1; const float* wd2; const float* bd2;
  const float* wt1; const float* bt1; const float* wt2; const float* bt2;
  const float* wv1; const float* bv1; const float* wv2; const float* bv2;
  const float* wv3; const float* bv3;
  const float* ws1; const float* bs1; const float* ws2; const float* bs2;
  const float* wp1; const float* bp1; const float* wp2; const float* bp2;
  const float* wp3; const float* bp3;
  float* out;
};

// fp32 (Kmem x N, row-major) global -> f16 column-major LDS.
__device__ inline void load_w(const float* w, _Float16* dst, int Kmem, int N,
                              int Kpad, int tid) {
  int total = Kmem * N;
  for (int i = tid; i < total; i += 64) {
    int k = i / N;
    int n = i - k * N;
    dst[n * Kpad + k] = (_Float16)w[i];
  }
}

// A fragment (16xK f16 row-major, row stride `stride` halves).
__device__ inline v16h load_a(const _Float16* A, int stride, int ks, int lane) {
  const _Float16* p = A + (lane & 15) * stride + ks * 32 + ((lane >> 4) & 1) * 8;
  v8h lo = *(const v8h*)(p);
  v8h hi = *(const v8h*)(p + 16);
  v16h r;
#pragma unroll
  for (int i = 0; i < 8; ++i) { r[i] = lo[i]; r[8 + i] = hi[i]; }
  return r;
}

// B fragment (32x16 tile) from column-major LDS (col stride = Kpad halves).
__device__ inline v16h load_b(const _Float16* W, int Kpad, int ks, int nt, int lane) {
  const _Float16* p = W + ((lane & 15) + nt * 16) * Kpad + ks * 32
                        + ((lane >> 4) & 1) * 8;
  v8h lo = *(const v8h*)(p);
  v8h hi = *(const v8h*)(p + 16);
  v16h r;
#pragma unroll
  for (int i = 0; i < 8; ++i) { r[i] = lo[i]; r[8 + i] = hi[i]; }
  return r;
}

template <int KS, int NT>
__device__ inline void gemm_relu(const _Float16* A, const _Float16* W,
                                 const float* bias, _Float16* dst, int dstStride,
                                 int lane, bool relu) {
  v16h a[KS];
#pragma unroll
  for (int ks = 0; ks < KS; ++ks) a[ks] = load_a(A, KS * 32, ks, lane);
  const int mb = (lane >> 4) * 8;
#pragma unroll
  for (int nt = 0; nt < NT; ++nt) {
    v8f acc = {};
#pragma unroll
    for (int ks = 0; ks < KS; ++ks)
      acc = WMMA(a[ks], load_b(W, KS * 32, ks, nt, lane), acc);
    const int n = (lane & 15) + nt * 16;
    const float bv = bias[n];
#pragma unroll
    for (int r = 0; r < 8; ++r) {
      float v = acc[r] + bv;
      if (relu) v = fmaxf(v, 0.0f);
      dst[(mb + r) * dstStride + n] = (_Float16)v;
    }
  }
}

template <int KS>
__device__ inline void gemm_col0(const _Float16* A, const _Float16* W,
                                 const float* bias, float* out16, int lane) {
  v8f acc = {};
#pragma unroll
  for (int ks = 0; ks < KS; ++ks)
    acc = WMMA(load_a(A, KS * 32, ks, lane), load_b(W, KS * 32, ks, 0, lane), acc);
  if ((lane & 15) == 0) {
    const int mb = (lane >> 4) * 8;
    const float bv = bias[0];
#pragma unroll
    for (int r = 0; r < 8; ++r) out16[mb + r] = acc[r] + bv;
  }
}

__device__ inline int clamp_pid(float f) {
  int p = (int)f;
  p = p < 0 ? 0 : p;
  return p > 449 ? 449 : p;
}

__global__ __launch_bounds__(64, 1)
void dueling_q_kernel(Params P) {
  __shared__ __align__(32) _Float16 WBUF[TOTW];
  __shared__ __align__(16) float    STATB[2][16 * 78];  // staged state tile
  __shared__ __align__(32) _Float16 BASEB[2][16 * 64];  // base, 53 -> pad 64
  __shared__ __align__(32) _Float16 HBUF[2][16 * 64];
  __shared__ __align__(32) _Float16 TBUF[2][16 * 32];
  __shared__ float VALS[2][16];
  __shared__ float ASH[2][16];
  __shared__ float APASS[2][4][16];

  const int tid = threadIdx.x;
  const int w = tid >> 5;
  const int lane = tid & 31;
  const float* state = P.state;

  // ---- zero LDS padding regions, then stage f16 weights once per block ----
  for (unsigned i = tid; i < TOTW; i += 64) WBUF[i] = (_Float16)0.0f;
  {
    _Float16* z = &BASEB[0][0];
    for (int i = tid; i < 2 * 16 * 64; i += 64) z[i] = (_Float16)0.0f;
    z = &HBUF[0][0];
    for (int i = tid; i < 2 * 16 * 64; i += 64) z[i] = (_Float16)0.0f;
    z = &TBUF[0][0];
    for (int i = tid; i < 2 * 16 * 32; i += 64) z[i] = (_Float16)0.0f;
  }
  __syncthreads();
  load_w(P.wd1, WBUF + OWD1,  4, 32, 32, tid);
  load_w(P.wd2, WBUF + OWD2, 32, 32, 32, tid);
  load_w(P.wt1, WBUF + OWT1, 18, 32, 32, tid);
  load_w(P.wt2, WBUF + OWT2, 32, 32, 32, tid);
  load_w(P.wv1, WBUF + OWV1, 53, 64, 64, tid);
  load_w(P.wv2, WBUF + OWV2, 64, 32, 64, tid);
  load_w(P.wv3, WBUF + OWV3, 32,  1, 32, tid);
  load_w(P.ws1, WBUF + OWS1, 53, 32, 64, tid);
  load_w(P.ws2, WBUF + OWS2, 32,  1, 32, tid);
  load_w(P.wp1,           WBUF + OWP1A, 32, 64, 32, tid);  // rows 0..31
  load_w(P.wp1 + 32 * 64, WBUF + OWP1B, 53, 64, 64, tid);  // rows 32..84
  load_w(P.wp2, WBUF + OWP2, 64, 16, 64, tid);
  load_w(P.wp3, WBUF + OWP3, 16,  1, 32, tid);
  __syncthreads();

  float* statb = STATB[w];
  _Float16* baseb = BASEB[w];
  _Float16* hbuf = HBUF[w];
  _Float16* tbuf = TBUF[w];
  const unsigned statb_lds = (unsigned)(size_t)statb;  // LDS byte offset
  const int wave_tile0 = (blockIdx.x * 2 + w) * TILES;

  for (int t = 0; t < TILES; ++t) {
    const int s0 = (wave_tile0 + t) * 16;

    // ---- async-stage this tile's 16 contiguous state rows (4992 B) --------
    {
      const float* gsrc = state + (size_t)s0 * 78;
#pragma unroll
      for (int it = 0; it < 10; ++it) {
        int idx = it * 32 + lane;                      // 312 b128 transfers
        if (idx < 312) {
          unsigned laddr = statb_lds + (unsigned)idx * 16u;
          unsigned long long gaddr =
              (unsigned long long)(size_t)(gsrc + (size_t)idx * 4);
          asm volatile("global_load_async_to_lds_b128 %0, %1, off"
                       :: "v"(laddr), "v"(gaddr) : "memory");
        }
      }
      if (t + 1 < TILES)  // pull next tile toward L2 while we compute
        __builtin_prefetch(
            (const void*)((const char*)(state + (size_t)(s0 + 16) * 78) +
                          (size_t)lane * 128),
            0, 0);
      asm volatile("s_wait_asynccnt 0x0" ::: "memory");
    }
    __syncthreads();

    // ================= defender MLP + mean pool =================
    v8f pool0 = {}, pool1 = {};
    for (int dt = 0; dt < 5; ++dt) {
      if (lane < 16) {
        const float* row = statb + lane * 78;
        _Float16* tt = tbuf + lane * 32;
        tt[0] = (_Float16)row[23 + 2 * dt];  // def_pos
        tt[1] = (_Float16)row[24 + 2 * dt];
        tt[2] = (_Float16)row[43 + 2 * dt];  // def_vel
        tt[3] = (_Float16)row[44 + 2 * dt];
#pragma unroll
        for (int c = 4; c < 16; ++c) tt[c] = (_Float16)0.0f;  // K padding
      }
      __syncthreads();
      gemm_relu<1, 2>(tbuf, WBUF + OWD1, P.bd1, hbuf, 32, lane, true);
      __syncthreads();
      {
        v16h a = load_a(hbuf, 32, 0, lane);
        v16h b0 = load_b(WBUF + OWD2, 32, 0, 0, lane);
        v16h b1 = load_b(WBUF + OWD2, 32, 0, 1, lane);
        v8f a0 = {}, a1 = {};
        a0 = WMMA(a, b0, a0);
        a1 = WMMA(a, b1, a1);
        const int n = lane & 15;
        const float bv0 = P.bd2[n], bv1 = P.bd2[n + 16];
#pragma unroll
        for (int r = 0; r < 8; ++r) {
          float x0 = a0[r] + bv0; pool0[r] += x0 > 0.f ? x0 : 0.f;
          float x1 = a1[r] + bv1; pool1[r] += x1 > 0.f ? x1 : 0.f;
        }
      }
      __syncthreads();
    }

    // ========= base = [context(11), bh_vel(2), bh_embed(8), pooled(32)] ====
    if (lane < 16) {
      const float* row = statb + lane * 78;
      _Float16* dst = baseb + lane * 64;
#pragma unroll
      for (int c = 0; c < 11; ++c) dst[c] = (_Float16)row[c];
      dst[11] = (_Float16)row[41];
      dst[12] = (_Float16)row[42];
      const float* e = P.emb + clamp_pid(row[73]) * 8;
#pragma unroll
      for (int k = 0; k < 8; ++k) dst[13 + k] = (_Float16)e[k];
    }
    {
      const int n = lane & 15, mb = (lane >> 4) * 8;
#pragma unroll
      for (int r = 0; r < 8; ++r) {
        baseb[(mb + r) * 64 + 21 + n] = (_Float16)(pool0[r] * 0.2f);
        baseb[(mb + r) * 64 + 37 + n] = (_Float16)(pool1[r] * 0.2f);
      }
    }
    __syncthreads();

    // ================= value & shoot heads =================
    gemm_relu<2, 4>(baseb, WBUF + OWV1, P.bv1, hbuf, 64, lane, true);
    __syncthreads();
    gemm_relu<2, 2>(hbuf, WBUF + OWV2, P.bv2, tbuf, 32, lane, true);
    __syncthreads();
    gemm_col0<1>(tbuf, WBUF + OWV3, P.bv3, VALS[w], lane);
    __syncthreads();
    gemm_relu<2, 2>(baseb, WBUF + OWS1, P.bs1, hbuf, 32, lane, true);
    __syncthreads();
    gemm_col0<1>(hbuf, WBUF + OWS2, P.bs2, ASH[w], lane);
    __syncthreads();
    for (int i = lane; i < 512; i += 32) tbuf[i] = (_Float16)0.0f;  // clean
    __syncthreads();

    // ================= teammates: phi net + fused pass head ================
    for (int j = 0; j < 4; ++j) {
      if (lane < 16) {
        const float* row = statb + lane * 78;
        _Float16* tt = tbuf + lane * 32;
        tt[0] = (_Float16)row[33 + 2 * j];  // tm_pos
        tt[1] = (_Float16)row[34 + 2 * j];
        tt[2] = (_Float16)row[53 + 2 * j];  // tm_vel
        tt[3] = (_Float16)row[54 + 2 * j];
        tt[4] = (_Float16)row[11 + j];      // tm_open
        tt[5] = (_Float16)row[15 + j];      // tm_fg
        tt[6] = (_Float16)row[19 + j];      // tm_dist
        tt[7] = (_Float16)row[61 + j];      // tm_lane_min
        tt[8] = (_Float16)row[65 + j];      // tm_corridor
        tt[9] = (_Float16)row[69 + j];      // tm_pass_dist
        const float* e = P.emb + clamp_pid(row[74 + j]) * 8;
#pragma unroll
        for (int k = 0; k < 8; ++k) tt[10 + k] = (_Float16)e[k];
      }
      __syncthreads();
      gemm_relu<1, 2>(tbuf, WBUF + OWT1, P.bt1, hbuf, 32, lane, true);
      __syncthreads();
      gemm_relu<1, 2>(hbuf, WBUF + OWT2, P.bt2, tbuf, 32, lane, true);  // phi
      __syncthreads();
      // p1 fused: p_in@wp1 = phi@wp1[0:32] + base@wp1[32:85]
      {
        v16h aphi = load_a(tbuf, 32, 0, lane);
        v16h ab0 = load_a(baseb, 64, 0, lane);
        v16h ab1 = load_a(baseb, 64, 1, lane);
        const int mb = (lane >> 4) * 8;
#pragma unroll
        for (int nt = 0; nt < 4; ++nt) {
          v8f acc = {};
          acc = WMMA(aphi, load_b(WBUF + OWP1A, 32, 0, nt, lane), acc);
          acc = WMMA(ab0,  load_b(WBUF + OWP1B, 64, 0, nt, lane), acc);
          acc = WMMA(ab1,  load_b(WBUF + OWP1B, 64, 1, nt, lane), acc);
          const int n = (lane & 15) + nt * 16;
          const float bv = P.bp1[n];
#pragma unroll
          for (int r = 0; r < 8; ++r) {
            float v = fmaxf(acc[r] + bv, 0.0f);
            hbuf[(mb + r) * 64 + n] = (_Float16)v;
          }
        }
      }
      __syncthreads();
      gemm_relu<2, 1>(hbuf, WBUF + OWP2, P.bp2, tbuf, 32, lane, true);
      // cols 16..31 hold stale phi values; clear K padding before p3.
      for (int i = lane; i < 256; i += 32)
        tbuf[(i >> 4) * 32 + 16 + (i & 15)] = (_Float16)0.0f;
      __syncthreads();
      gemm_col0<1>(tbuf, WBUF + OWP3, P.bp3, APASS[w][j], lane);
      __syncthreads();
    }

    // ================= combine: q = value + advantages =================
    if (lane < 16) {
      const float v = VALS[w][lane];
      const float a0 = APASS[w][0][lane], a1 = APASS[w][1][lane];
      const float a2 = APASS[w][2][lane], a3 = APASS[w][3][lane];
      const float mean = 0.25f * (a0 + a1 + a2 + a3);
      float* o = P.out + (size_t)(s0 + lane) * 5;
      o[0] = v + ASH[w][lane];
      o[1] = v + a0 - mean;
      o[2] = v + a1 - mean;
      o[3] = v + a2 - mean;
      o[4] = v + a3 - mean;
    }
    __syncthreads();
  }
}

extern "C" void kernel_launch(void* const* d_in, const int* in_sizes, int n_in,
                              void* d_out, int out_size, void* d_ws, size_t ws_size,
                              hipStream_t stream) {
  (void)n_in; (void)out_size; (void)d_ws; (void)ws_size;
  Params P;
  P.state = (const float*)d_in[0];  P.emb = (const float*)d_in[1];
  P.wd1 = (const float*)d_in[2];    P.bd1 = (const float*)d_in[3];
  P.wd2 = (const float*)d_in[4];    P.bd2 = (const float*)d_in[5];
  P.wt1 = (const float*)d_in[6];    P.bt1 = (const float*)d_in[7];
  P.wt2 = (const float*)d_in[8];    P.bt2 = (const float*)d_in[9];
  P.wv1 = (const float*)d_in[10];   P.bv1 = (const float*)d_in[11];
  P.wv2 = (const float*)d_in[12];   P.bv2 = (const float*)d_in[13];
  P.wv3 = (const float*)d_in[14];   P.bv3 = (const float*)d_in[15];
  P.ws1 = (const float*)d_in[16];   P.bs1 = (const float*)d_in[17];
  P.ws2 = (const float*)d_in[18];   P.bs2 = (const float*)d_in[19];
  P.wp1 = (const float*)d_in[20];   P.bp1 = (const float*)d_in[21];
  P.wp2 = (const float*)d_in[22];   P.bp2 = (const float*)d_in[23];
  P.wp3 = (const float*)d_in[24];   P.bp3 = (const float*)d_in[25];
  P.out = (float*)d_out;

  const int B = in_sizes[0] / 78;                 // 262144
  const int blocks = B / (32 * TILES);            // 2 waves x 8 tiles x 16
  dueling_q_kernel<<<blocks, 64, 0, stream>>>(P);
}
